// EncoderBlock_21938692948518
// MI455X (gfx1250) — matrix operand order repaired and checked by reading, not compiled, not run on hardware
//
#include <hip/hip_runtime.h>
#include <hip/hip_bf16.h>

typedef __attribute__((ext_vector_type(16))) __bf16 v16bf;
typedef __attribute__((ext_vector_type(8)))  float  v8f;
typedef __attribute__((ext_vector_type(4)))  int    v4i_t;

#define LDT 40      // padded bf16 row stride for GEMM LDS tiles (80B, 16B aligned, conflict-spread)
#define AT_LD 72    // padded bf16 row stride for attention LDS tiles (144B)

#if defined(__HIP_DEVICE_COMPILE__) && __has_builtin(__builtin_amdgcn_global_load_async_to_lds_b128)
#define HAVE_ASYNC_LDS 1
#else
#define HAVE_ASYNC_LDS 0
#endif

// 16-byte global -> LDS copy. On CDNA5 use the async LDS-DMA path (no VGPR
// round-trip, tracked by ASYNCcnt, overlaps with WMMA); else vector copy.
__device__ __forceinline__ void copy16_g2l(const __bf16* __restrict__ g, __bf16* l) {
#if HAVE_ASYNC_LDS
    __builtin_amdgcn_global_load_async_to_lds_b128(
        (__attribute__((address_space(1))) v4i_t*)(v4i_t*)(void*)g,
        (__attribute__((address_space(3))) v4i_t*)(v4i_t*)(void*)l, 0, 0);
#else
    *(uint4*)l = *(const uint4*)g;
#endif
}

__device__ __forceinline__ void async_join() {
#if HAVE_ASYNC_LDS
#if __has_builtin(__builtin_amdgcn_s_wait_asynccnt)
    __builtin_amdgcn_s_wait_asynccnt(0);
#else
    asm volatile("s_wait_asynccnt 0" ::: "memory");
#endif
#endif
}

__device__ __forceinline__ v8f v8f_zero() {
    v8f z;
#pragma unroll
    for (int i = 0; i < 8; ++i) z[i] = 0.f;
    return z;
}

// A fragment: 16x32 bf16, lane m=lane&15 holds row m.
// lanes 0-15: elems 0..7 -> K=0..7, elems 8..15 -> K=16..23
// lanes 16-31: elems 0..7 -> K=8..15, elems 8..15 -> K=24..31
__device__ __forceinline__ v16bf load_a_frag(const __bf16* base, int ld, int lane) {
    const int m = lane & 15, half = lane >> 4;
    const __bf16* p = base + m * ld + half * 8;
    v16bf a;
#pragma unroll
    for (int e = 0; e < 8; ++e) { a[e] = p[e]; a[e + 8] = p[e + 16]; }
    return a;
}

// B fragment: 32x16 bf16 stored K-contiguous per column: lane n=lane&15 is column,
// elems 0..7 -> K = (lane>>4)*8 + e, elems 8..15 -> K = 16 + (lane>>4)*8 + (e-8). `base` points at [col][K] with stride ld.
__device__ __forceinline__ v16bf load_b_frag(const __bf16* base, int ld, int lane) {
    const int n = lane & 15, kb = (lane >> 4) * 8;
    const __bf16* p = base + n * ld + kb;
    v16bf b;
#pragma unroll
    for (int e = 0; e < 8; ++e) { b[e] = p[e]; b[e + 8] = p[e + 16]; }
    return b;
}

// ---------------------------------------------------------------------------
// One-time weight prep: W(f32,[K][N]) -> Wt(bf16,[N][K]) via 64x64 LDS tiles.
// Coalesced f32 reads, 16B bf16 writes along K.
// ---------------------------------------------------------------------------
__global__ __launch_bounds__(256) void wconvert_kernel(const float* __restrict__ W,
                                                       __bf16* __restrict__ Wt,
                                                       int K, int N) {
    __shared__ __align__(16) __bf16 t[64][AT_LD];
    const int tid = threadIdx.x;
    const int k0 = blockIdx.y * 64, n0 = blockIdx.x * 64;
    const int c = tid & 63, r4 = tid >> 6;
#pragma unroll
    for (int p = 0; p < 16; ++p) {
        int r = p * 4 + r4;
        t[c][r] = (__bf16)W[(size_t)(k0 + r) * N + n0 + c];
    }
    __syncthreads();
#pragma unroll
    for (int i = 0; i < 2; ++i) {
        int e = tid + i * 256;               // 512 chunks of 8 bf16
        int c2 = e >> 3, kc = (e & 7) * 8;
        *(uint4*)&Wt[(size_t)(n0 + c2) * K + k0 + kc] = *(const uint4*)&t[c2][kc];
    }
}

// ---------------------------------------------------------------------------
// LayerNorm (ddof=1, eps added to std, scalar alpha/bias) -> bf16 output
// ---------------------------------------------------------------------------
__global__ __launch_bounds__(256) void ln_kernel(const float* __restrict__ x,
                                                 const float* __restrict__ alpha,
                                                 const float* __restrict__ beta,
                                                 __bf16* __restrict__ out, int D) {
    __shared__ float red[16];
    const int row = blockIdx.x, tid = threadIdx.x;
    const float* xr = x + (size_t)row * D;
    float s = 0.f, ss = 0.f;
    for (int i = tid; i < D; i += 256) { float v = xr[i]; s += v; ss += v * v; }
#pragma unroll
    for (int o = 16; o > 0; o >>= 1) { s += __shfl_xor(s, o, 32); ss += __shfl_xor(ss, o, 32); }
    if ((tid & 31) == 0) { red[tid >> 5] = s; red[8 + (tid >> 5)] = ss; }
    __syncthreads();
    if (tid == 0) {
        float a = 0.f, b = 0.f;
#pragma unroll
        for (int i = 0; i < 8; ++i) { a += red[i]; b += red[8 + i]; }
        red[0] = a; red[1] = b;
    }
    __syncthreads();
    const float mean = red[0] / (float)D;
    const float var  = (red[1] - (float)D * mean * mean) / (float)(D - 1);
    const float inv  = alpha[0] / (sqrtf(fmaxf(var, 0.f)) + 1e-6f);
    const float bb   = beta[0];
    __bf16* orow = out + (size_t)row * D;
    for (int i = tid; i < D; i += 256)
        orow[i] = (__bf16)((xr[i] - mean) * inv + bb);
}

// ---------------------------------------------------------------------------
// WMMA GEMM: C[M,N] = A(bf16,[M][K]) * Wt(bf16,[N][K]) + bias
//   optional ReLU, optional residual(f32), outputs f32 and/or bf16.
// 128x128 block, BK=32, 8 waves (4 row-groups x 2 col-groups),
// double-buffered LDS filled by async global->LDS DMA.
// ---------------------------------------------------------------------------
__global__ __launch_bounds__(256) void gemm_bf16_kernel(
    const __bf16* __restrict__ A, const __bf16* __restrict__ Wt,
    const float* __restrict__ bias, const float* __restrict__ res,
    float* __restrict__ outF, __bf16* __restrict__ outB,
    int M, int K, int N, int relu) {
    __shared__ __align__(16) __bf16 As[2][128 * LDT];
    __shared__ __align__(16) __bf16 Bs[2][128 * LDT];

    const int tid = threadIdx.x;
    const int lane = tid & 31, w = tid >> 5;
    const int wr = (w & 3) * 32;   // wave row offset in block tile
    const int wc = (w >> 2) * 64;  // wave col offset in block tile
    const int row0 = blockIdx.y * 128, col0 = blockIdx.x * 128;
    const int nk = K >> 5;

    v8f acc[2][4];
#pragma unroll
    for (int i = 0; i < 2; ++i)
#pragma unroll
        for (int j = 0; j < 4; ++j) acc[i][j] = v8f_zero();

    auto stage = [&](int buf, int k0) {
        // A and Wt tiles: 128 rows x 32 K bf16 each, 16B async chunks
#pragma unroll
        for (int i = 0; i < 2; ++i) {
            int e = tid + i * 256;              // 512 chunks of 8 bf16
            int r = e >> 2, kc = (e & 3) * 8;
            copy16_g2l(A  + (size_t)(row0 + r) * K + k0 + kc, &As[buf][r * LDT + kc]);
            copy16_g2l(Wt + (size_t)(col0 + r) * K + k0 + kc, &Bs[buf][r * LDT + kc]);
        }
        if (k0 + 64 < K)  // prefetch one tile ahead of the double buffer
            __builtin_prefetch(&Wt[(size_t)(col0 + (tid & 127)) * K + k0 + 64], 0, 1);
    };

    stage(0, 0);
    async_join();
    __syncthreads();
    for (int kt = 0; kt < nk; ++kt) {
        const int cur = kt & 1;
        if (kt + 1 < nk) stage(cur ^ 1, (kt + 1) * 32);
        const __bf16* as = As[cur];
        const __bf16* bs = Bs[cur];
        v16bf af[2], bf[4];
#pragma unroll
        for (int i = 0; i < 2; ++i) af[i] = load_a_frag(as + (wr + i * 16) * LDT, LDT, lane);
#pragma unroll
        for (int j = 0; j < 4; ++j) bf[j] = load_b_frag(bs + (wc + j * 16) * LDT, LDT, lane);
#pragma unroll
        for (int i = 0; i < 2; ++i)
#pragma unroll
            for (int j = 0; j < 4; ++j)
                acc[i][j] = __builtin_amdgcn_wmma_f32_16x16x32_bf16(
                    false, af[i], false, bf[j], (short)0, acc[i][j], false, false);
        async_join();
        __syncthreads();
    }

    const int m8 = (lane >> 4) * 8, nn = lane & 15;
#pragma unroll
    for (int i = 0; i < 2; ++i) {
#pragma unroll
        for (int j = 0; j < 4; ++j) {
            const int col = col0 + wc + j * 16 + nn;
            const float bv = bias[col];
#pragma unroll
            for (int r = 0; r < 8; ++r) {
                const int row = row0 + wr + i * 16 + m8 + r;
                float v = acc[i][j][r] + bv;
                if (relu) v = v > 0.f ? v : 0.f;
                if (res)  v += res[(size_t)row * N + col];
                if (outF) outF[(size_t)row * N + col] = v;
                if (outB) outB[(size_t)row * N + col] = (__bf16)v;
            }
        }
    }
}

// ---------------------------------------------------------------------------
// Flash attention: per wave a 16-row Q strip over full S with online softmax.
// KV chunks of 64; K staged (async) row-major, V staged transposed.
// grid = (B*H, S/128), block = 256 (8 waves).
// ---------------------------------------------------------------------------
__global__ __launch_bounds__(256) void attn_kernel(
    const __bf16* __restrict__ q, const __bf16* __restrict__ k,
    const __bf16* __restrict__ v, const int* __restrict__ mask,
    __bf16* __restrict__ ctx, int S, int D, int H) {
    __shared__ __align__(16) __bf16 Ks[64 * AT_LD];       // [key][dk]
    __shared__ __align__(16) __bf16 Vt[64 * AT_LD];       // [dk][key]
    __shared__ __align__(16) __bf16 Pb[8][16 * AT_LD];    // per-wave P tile [row][key]

    const int tid = threadIdx.x, lane = tid & 31, w = tid >> 5;
    const int b = blockIdx.x / H, h = blockIdx.x % H;
    const int q0 = blockIdx.y * 128 + w * 16;
    const int m = lane & 15, half = lane >> 4;
    const size_t base = ((size_t)b * S) * D + (size_t)h * 64;
    const int mbase = b * S;

    // Q fragments for dk chunks [0,32) and [32,64)
    v16bf aq[2];
#pragma unroll
    for (int c = 0; c < 2; ++c) {
        const __bf16* p = q + base + (size_t)(q0 + m) * D + c * 32 + half * 8;
#pragma unroll
        for (int e = 0; e < 8; ++e) { aq[c][e] = p[e]; aq[c][e + 8] = p[e + 16]; }
    }

    v8f cacc[4];
#pragma unroll
    for (int d = 0; d < 4; ++d) cacc[d] = v8f_zero();
    float mrow[8], lrow[8];
#pragma unroll
    for (int r = 0; r < 8; ++r) { mrow[r] = -3.0e38f; lrow[r] = 0.f; }

    for (int kv = 0; kv < S; kv += 64) {
        __syncthreads();
        // stage K tile (64 keys x 64 dk) via async LDS DMA
#pragma unroll
        for (int i = 0; i < 2; ++i) {
            int e = tid + i * 256;
            int key = e >> 3, dkc = (e & 7) * 8;
            copy16_g2l(k + base + (size_t)(kv + key) * D + dkc, &Ks[key * AT_LD + dkc]);
        }
        // stage V transposed (64 dk x 64 keys)
#pragma unroll
        for (int p2 = 0; p2 < 16; ++p2) {
            int idx = p2 * 256 + tid;
            int dk = idx & 63, key = idx >> 6;
            Vt[dk * AT_LD + key] = v[base + (size_t)(kv + key) * D + dk];
        }
        async_join();
        __syncthreads();

        // scores: 4 tiles of 16 keys, full DK=64 via two k=32 WMMAs
        float sc[4][8], cmax[8];
#pragma unroll
        for (int r = 0; r < 8; ++r) cmax[r] = -3.0e38f;
#pragma unroll
        for (int j = 0; j < 4; ++j) {
            v8f s = v8f_zero();
#pragma unroll
            for (int c = 0; c < 2; ++c) {
                v16bf bk = load_b_frag(Ks + (j * 16) * AT_LD + c * 32, AT_LD, lane);
                s = __builtin_amdgcn_wmma_f32_16x16x32_bf16(
                    false, aq[c], false, bk, (short)0, s, false, false);
            }
            const int mk = mask[mbase + kv + j * 16 + m];   // lane's key column
#pragma unroll
            for (int r = 0; r < 8; ++r) {
                float sv = mk ? s[r] * 0.125f : -1.0e9f;    // 1/sqrt(64)
                sc[j][r] = sv;
                cmax[r] = fmaxf(cmax[r], sv);
            }
        }
        // cross-lane row max within 16-lane groups, online rescale
#pragma unroll
        for (int r = 0; r < 8; ++r) {
            float mx = cmax[r];
#pragma unroll
            for (int o = 1; o < 16; o <<= 1) mx = fmaxf(mx, __shfl_xor(mx, o, 32));
            const float mnew = fmaxf(mrow[r], mx);
            const float fac = __expf(mrow[r] - mnew);
            mrow[r] = mnew;
            lrow[r] *= fac;
#pragma unroll
            for (int d = 0; d < 4; ++d) cacc[d][r] *= fac;
        }
        // P = exp(s - m), accumulate row sums, spill P to LDS (layout swap C->A)
        float prow[8];
#pragma unroll
        for (int r = 0; r < 8; ++r) prow[r] = 0.f;
#pragma unroll
        for (int j = 0; j < 4; ++j)
#pragma unroll
            for (int r = 0; r < 8; ++r) {
                float pv = __expf(sc[j][r] - mrow[r]);
                prow[r] += pv;
                Pb[w][(half * 8 + r) * AT_LD + j * 16 + m] = (__bf16)pv;
            }
#pragma unroll
        for (int r = 0; r < 8; ++r) {
            float ps = prow[r];
#pragma unroll
            for (int o = 1; o < 16; o <<= 1) ps += __shfl_xor(ps, o, 32);
            lrow[r] += ps;
        }
        // ctx += P(16x64) @ V(64x64): A from per-wave P LDS, B from Vt
#pragma unroll
        for (int c = 0; c < 2; ++c) {
            v16bf ap = load_a_frag(&Pb[w][c * 32], AT_LD, lane);
#pragma unroll
            for (int d = 0; d < 4; ++d) {
                v16bf bv = load_b_frag(Vt + (d * 16) * AT_LD + c * 32, AT_LD, lane);
                cacc[d] = __builtin_amdgcn_wmma_f32_16x16x32_bf16(
                    false, ap, false, bv, (short)0, cacc[d], false, false);
            }
        }
    }

    // normalize and store bf16 ctx in [B,S,H*DK] layout
#pragma unroll
    for (int r = 0; r < 8; ++r) {
        const float inv = 1.f / lrow[r];
        const int row = q0 + half * 8 + r;
#pragma unroll
        for (int d = 0; d < 4; ++d)
            ctx[base + (size_t)row * D + d * 16 + m] = (__bf16)(cacc[d][r] * inv);
    }
}

// ---------------------------------------------------------------------------
extern "C" void kernel_launch(void* const* d_in, const int* in_sizes, int n_in,
                              void* d_out, int out_size, void* d_ws, size_t ws_size,
                              hipStream_t stream) {
    (void)in_sizes; (void)n_in; (void)out_size; (void)ws_size;
    const int B = 2, S = 2048, D = 1024, H = 16, DFF = 4096;
    const int M = B * S;

    const float* x    = (const float*)d_in[0];
    const int*   mask = (const int*)d_in[1];
    const float* wq = (const float*)d_in[2];  const float* bq = (const float*)d_in[3];
    const float* wk = (const float*)d_in[4];  const float* bk = (const float*)d_in[5];
    const float* wv = (const float*)d_in[6];  const float* bv = (const float*)d_in[7];
    const float* wo = (const float*)d_in[8];  const float* bo = (const float*)d_in[9];
    const float* w1 = (const float*)d_in[10]; const float* b1 = (const float*)d_in[11];
    const float* w2 = (const float*)d_in[12]; const float* b2 = (const float*)d_in[13];
    const float* a1 = (const float*)d_in[14]; const float* g1 = (const float*)d_in[15];
    const float* a2 = (const float*)d_in[16]; const float* g2 = (const float*)d_in[17];

    char* ws = (char*)d_ws;
    const size_t MB = 1024ull * 1024ull;
    __bf16* n1  = (__bf16*)(ws + 0 * MB);     // B*S*D bf16  = 8 MB
    __bf16* qb  = (__bf16*)(ws + 8 * MB);     // 8 MB
    __bf16* kb  = (__bf16*)(ws + 16 * MB);    // 8 MB
    __bf16* vb  = (__bf16*)(ws + 24 * MB);    // 8 MB
    __bf16* cb  = (__bf16*)(ws + 32 * MB);    // 8 MB (attention context)
    float*  x1  = (float*) (ws + 40 * MB);    // B*S*D f32   = 16 MB
    __bf16* n2  = (__bf16*)(ws + 56 * MB);    // 8 MB
    __bf16* h1  = (__bf16*)(ws + 64 * MB);    // B*S*DFF bf16 = 32 MB
    __bf16* wqT = (__bf16*)(ws + 96 * MB);    // D*D bf16 = 2 MB each, transposed [N][K]
    __bf16* wkT = (__bf16*)(ws + 98 * MB);
    __bf16* wvT = (__bf16*)(ws + 100 * MB);
    __bf16* woT = (__bf16*)(ws + 102 * MB);
    __bf16* w1T = (__bf16*)(ws + 104 * MB);   // DFF*D bf16 = 8 MB  [DFF][D]
    __bf16* w2T = (__bf16*)(ws + 112 * MB);   // D*DFF bf16 = 8 MB  [D][DFF] (end: 120 MB)

    // 0) one-time weight convert+transpose to bf16 [N][K]
    wconvert_kernel<<<dim3(D / 64, D / 64), 256, 0, stream>>>(wq, wqT, D, D);
    wconvert_kernel<<<dim3(D / 64, D / 64), 256, 0, stream>>>(wk, wkT, D, D);
    wconvert_kernel<<<dim3(D / 64, D / 64), 256, 0, stream>>>(wv, wvT, D, D);
    wconvert_kernel<<<dim3(D / 64, D / 64), 256, 0, stream>>>(wo, woT, D, D);
    wconvert_kernel<<<dim3(DFF / 64, D / 64), 256, 0, stream>>>(w1, w1T, D, DFF);
    wconvert_kernel<<<dim3(D / 64, DFF / 64), 256, 0, stream>>>(w2, w2T, DFF, D);

    // 1) pre-norm 1
    ln_kernel<<<M, 256, 0, stream>>>(x, a1, g1, n1, D);
    // 2) QKV projections (bf16 outputs for attention)
    gemm_bf16_kernel<<<dim3(D / 128, M / 128), 256, 0, stream>>>(
        n1, wqT, bq, nullptr, nullptr, qb, M, D, D, 0);
    gemm_bf16_kernel<<<dim3(D / 128, M / 128), 256, 0, stream>>>(
        n1, wkT, bk, nullptr, nullptr, kb, M, D, D, 0);
    gemm_bf16_kernel<<<dim3(D / 128, M / 128), 256, 0, stream>>>(
        n1, wvT, bv, nullptr, nullptr, vb, M, D, D, 0);
    // 3) attention
    attn_kernel<<<dim3(B * H, S / 128), 256, 0, stream>>>(qb, kb, vb, mask, cb, S, D, H);
    // 4) output projection + residual -> x1 (f32)
    gemm_bf16_kernel<<<dim3(D / 128, M / 128), 256, 0, stream>>>(
        cb, woT, bo, x, x1, nullptr, M, D, D, 0);
    // 5) pre-norm 2
    ln_kernel<<<M, 256, 0, stream>>>(x1, a2, g2, n2, D);
    // 6) FFN up + ReLU (bf16)
    gemm_bf16_kernel<<<dim3(DFF / 128, M / 128), 256, 0, stream>>>(
        n2, w1T, b1, nullptr, nullptr, h1, M, D, DFF, 1);
    // 7) FFN down + residual -> final output (f32)
    gemm_bf16_kernel<<<dim3(D / 128, M / 128), 256, 0, stream>>>(
        h1, w2T, b2, x1, (float*)d_out, nullptr, M, DFF, D, 0);
}
